// PoseDecoderSparse_69114613727361
// MI455X (gfx1250) — matrix operand order repaired and checked by reading, not compiled, not run on hardware
//
#include <hip/hip_runtime.h>
#include <hip/hip_bf16.h>

#define TAU_INV 5.0f
#define EPSC 1e-8f

typedef __attribute__((ext_vector_type(16))) _Float16 v16h;
typedef __attribute__((ext_vector_type(8)))  _Float16 v8h;
typedef __attribute__((ext_vector_type(8)))  float    v8f;
typedef __attribute__((ext_vector_type(2)))  _Float16 h2v;

#define BT_ROWS 32
#define BT_PITCH 264              // 256 + 8 halfs pad -> 528B row stride (bank-conflict-free)
#define BT_BYTES (BT_ROWS * BT_PITCH * 2)

// ---------------- WMMA A-fragment loader (wave32, 16x16x32 f16) ----------------
// A (16x32, MxK): lane L: m=L&15; lanes16-31 shift K by +16; vgpr p -> K=2p
__device__ __forceinline__ v16h load_fragA(const _Float16* base, int row0, int lane, int k0, int C) {
  int m  = lane & 15;
  int hi = lane >> 4;
  const _Float16* rp = base + (size_t)(row0 + m) * C + k0 + hi * 16;
  v16h a;
#pragma unroll
  for (int p = 0; p < 8; ++p) {
    int kk = p << 1;
    h2v h = *(const h2v*)(rp + kk);
    a[2 * p] = h.x; a[2 * p + 1] = h.y;
  }
  return a;
}

// ---------------- CDNA5 async DMA: global -> LDS, ASYNCcnt tracked ----------------
__device__ __forceinline__ void async_b128(uint32_t lds_off, uint64_t gaddr) {
  asm volatile("global_load_async_to_lds_b128 %0, %1, off"
               :: "v"(lds_off), "v"(gaddr) : "memory");
}
__device__ __forceinline__ void wait_async0() {
  asm volatile("s_wait_asynccnt 0x0" ::: "memory");
}
// stage 32 rows x 256 halfs (16KB) of B (row-major, stride C halfs) into LDS buffer
__device__ __forceinline__ void stage_btile(const _Float16* fB, int j0, int C,
                                            uint32_t lbase, int tid) {
  const char* g = (const char*)(fB + (size_t)j0 * C);
  int ch = tid;                  // 1024 16-byte chunks, 4 per thread
#pragma unroll
  for (int k = 0; k < 4; ++k, ch += 256) {
    int row = ch >> 5;
    int cb  = (ch & 31) << 4;    // byte offset within row
    async_b128(lbase + (uint32_t)(row * (BT_PITCH * 2) + cb),
               (uint64_t)(uintptr_t)(g + (size_t)row * C * 2 + cb));
  }
}
// B fragment from LDS: lane L: n=L&15 picks row (column of sim), hi*16 K offset, 16 consecutive halfs
__device__ __forceinline__ v16h load_fragB_lds(const _Float16* bp, int ks) {
  v8h lo = *(const v8h*)(bp + ks * 32);
  v8h hi = *(const v8h*)(bp + ks * 32 + 8);
  return __builtin_shufflevector(lo, hi, 0,1,2,3,4,5,6,7,8,9,10,11,12,13,14,15);
}

__device__ __forceinline__ float red16(float v) {  // reduce within 16-lane half-wave
  v += __shfl_xor(v, 1);  v += __shfl_xor(v, 2);
  v += __shfl_xor(v, 4);  v += __shfl_xor(v, 8);
  return v;
}
__device__ __forceinline__ float red32(float v) {  // full wave32 reduce
  v = red16(v); v += __shfl_xor(v, 16);
  return v;
}

// software-pipelined 8-step K loop: load ks+1 while WMMA consumes ks
__device__ __forceinline__ v8f kloop_wmma(const v16h afr[8], const _Float16* bp) {
  v16h bcur = load_fragB_lds(bp, 0);
  v8f c = {};
#pragma unroll
  for (int ks = 0; ks < 7; ++ks) {
    v16h bnext = load_fragB_lds(bp, ks + 1);
    c = __builtin_amdgcn_wmma_f32_16x16x32_f16(false, afr[ks], false, bcur,
                                               (short)0, c, false, false);
    bcur = bnext;
  }
  c = __builtin_amdgcn_wmma_f32_16x16x32_f16(false, afr[7], false, bcur,
                                             (short)0, c, false, false);
  return c;
}

// ---------------- utility kernels ----------------
__global__ void zero_kernel(float* p, size_t n) {
  size_t i = (size_t)blockIdx.x * blockDim.x + threadIdx.x;
  size_t s = (size_t)gridDim.x * blockDim.x;
  for (; i < n; i += s) p[i] = 0.f;
}
__global__ void cvt_kernel(const float* f, _Float16* h, size_t n) {
  size_t i = (size_t)blockIdx.x * blockDim.x + threadIdx.x;
  size_t s = (size_t)gridDim.x * blockDim.x;
  for (; i < n; i += s) h[i] = (_Float16)f[i];
}

// ---------------- pass 1: rowsum(exp(sim/tau)) (swap=1 gives colsum) ----------------
__global__ void sumexp_kernel(const _Float16* hfeats, float* out, int N, int C, int swap) {
  __shared__ __align__(16) _Float16 btile[2][BT_ROWS][BT_PITCH];
  int b = blockIdx.y;
  const _Float16* fA = hfeats + (size_t)(2 * b + swap) * N * C;
  const _Float16* fB = hfeats + (size_t)(2 * b + 1 - swap) * N * C;
  float* rs = out + (size_t)b * N;
  int lane = threadIdx.x & 31;
  int wave = threadIdx.x >> 5;
  int wm = wave >> 1, wn = wave & 1;
  int i0 = blockIdx.x * 64 + wm * 16;
  int hi = lane >> 4;
  uint32_t lbase = (uint32_t)(uintptr_t)(&btile[0][0][0]);

  v16h afr[8];
#pragma unroll
  for (int ks = 0; ks < 8; ++ks) afr[ks] = load_fragA(fA, i0, lane, ks * 32, C);

  float acc[8];
#pragma unroll
  for (int r = 0; r < 8; ++r) acc[r] = 0.f;

  int nj = N >> 5;
  stage_btile(fB, 0, C, lbase, threadIdx.x);
  for (int jt = 0; jt < nj; ++jt) {
    int buf = jt & 1;
    wait_async0();
    __syncthreads();
    if (jt + 1 < nj)
      stage_btile(fB, (jt + 1) * 32, C, lbase + (uint32_t)((buf ^ 1) * BT_BYTES), threadIdx.x);
    const _Float16* bp = &btile[buf][wn * 16 + (lane & 15)][hi * 16];
    v8f c = kloop_wmma(afr, bp);
#pragma unroll
    for (int r = 0; r < 8; ++r) acc[r] += __expf(c[r] * TAU_INV);
  }
#pragma unroll
  for (int r = 0; r < 8; ++r) {
    float v = red16(acc[r]);
    if ((lane & 15) == 0) atomicAdd(&rs[i0 + hi * 8 + r], v);
  }
}

// ---------------- pass 2: M-weighted reductions ----------------
__global__ void pass2_kernel(const _Float16* hfeats, const float* rowsum, const float* colsum,
                             const float* scores, const float* kpts,
                             float* Mrow, float* Mxy, float* Ms1, float* sumMS,
                             int N, int C) {
  __shared__ __align__(16) _Float16 btile[2][BT_ROWS][BT_PITCH];
  int b = blockIdx.y;
  const _Float16* fA = hfeats + (size_t)(2 * b) * N * C;
  const _Float16* fB = hfeats + (size_t)(2 * b + 1) * N * C;
  const float* rsv = rowsum + (size_t)b * N;
  const float* csv = colsum + (size_t)b * N;
  const float* s1  = scores + (size_t)(2 * b + 1) * N;
  const float* xy1 = kpts   + (size_t)(2 * b + 1) * N * 2;
  float* mrow = Mrow + (size_t)b * N;
  float* mxy  = Mxy  + (size_t)b * N * 2;
  float* ms1  = Ms1  + (size_t)b * N;

  int lane = threadIdx.x & 31;
  int wave = threadIdx.x >> 5;
  int wm = wave >> 1, wn = wave & 1;
  int i0 = blockIdx.x * 64 + wm * 16;
  int hi = lane >> 4;
  uint32_t lbase = (uint32_t)(uintptr_t)(&btile[0][0][0]);

  v16h afr[8];
#pragma unroll
  for (int ks = 0; ks < 8; ++ks) afr[ks] = load_fragA(fA, i0, lane, ks * 32, C);

  float invr[8];
#pragma unroll
  for (int r = 0; r < 8; ++r) invr[r] = 1.0f / rsv[i0 + hi * 8 + r];

  float accM[8], accx[8], accy[8], accs[8], accms = 0.f;
#pragma unroll
  for (int r = 0; r < 8; ++r) { accM[r] = accx[r] = accy[r] = accs[r] = 0.f; }

  int nj = N >> 5;
  stage_btile(fB, 0, C, lbase, threadIdx.x);
  for (int jt = 0; jt < nj; ++jt) {
    int buf = jt & 1;
    int jc = jt * 32 + wn * 16 + (lane & 15);
    float ic  = 1.0f / csv[jc];
    float xx  = xy1[2 * jc];
    float yy  = xy1[2 * jc + 1];
    float ssc = s1[jc];
    wait_async0();
    __syncthreads();
    if (jt + 1 < nj)
      stage_btile(fB, (jt + 1) * 32, C, lbase + (uint32_t)((buf ^ 1) * BT_BYTES), threadIdx.x);
    const _Float16* bp = &btile[buf][wn * 16 + (lane & 15)][hi * 16];
    v8f c = kloop_wmma(afr, bp);
#pragma unroll
    for (int r = 0; r < 8; ++r) {
      float s = c[r];
      float e = __expf(s * TAU_INV);
      float M = e * e * invr[r] * ic;
      accM[r] += M;  accx[r] += M * xx;  accy[r] += M * yy;  accs[r] += M * ssc;
      accms   += M * s;
    }
  }
#pragma unroll
  for (int r = 0; r < 8; ++r) {
    float vM = red16(accM[r]);
    float vx = red16(accx[r]);
    float vy = red16(accy[r]);
    float vs = red16(accs[r]);
    if ((lane & 15) == 0) {
      int row = i0 + hi * 8 + r;
      atomicAdd(&mrow[row], vM);
      atomicAdd(&mxy[2 * row], vx);
      atomicAdd(&mxy[2 * row + 1], vy);
      atomicAdd(&ms1[row], vs);
    }
  }
  float vms = red32(accms);
  if (lane == 0) atomicAdd(&sumMS[b], vms);
}

// ---------------- pass 3: per-row weights -> AtA (9x9 sym, 45 entries) + stats ----------------
__global__ void stats_kernel(const float* Mrow, const float* Mxy, const float* Ms1,
                             const float* scores, const float* kpts, const float* Kmat,
                             float* AtA, float* misc, int N) {
  int b = blockIdx.x;
  const float* mrow = Mrow + (size_t)b * N;
  const float* mxy  = Mxy  + (size_t)b * N * 2;
  const float* ms1  = Ms1  + (size_t)b * N;
  const float* s0   = scores + (size_t)(2 * b) * N;
  const float* xy0  = kpts   + (size_t)(2 * b) * N * 2;
  const float* K0 = Kmat + (size_t)b * 18;
  const float* K1 = K0 + 9;
  float fx0 = K0[0], fy0 = K0[4], cx0 = K0[2], cy0 = K0[5];
  float fx1 = K1[0], fy1 = K1[4], cx1 = K1[2], cy1 = K1[5];

  float ata[45];
#pragma unroll
  for (int k = 0; k < 45; ++k) ata[k] = 0.f;
  float sumRS = 0.f, sumM = 0.f, nkeep = 0.f;

  for (int i = threadIdx.x; i < N; i += blockDim.x) {
    float rsu = mrow[i];
    float rsc = fmaxf(rsu, EPSC);
    float w = rsc * s0[i] * ms1[i];
    if (w > 0.f) nkeep += 1.f; else w = 0.f;
    float x1e = mxy[2 * i] / rsc, y1e = mxy[2 * i + 1] / rsc;
    float h0[3] = { (xy0[2 * i] - cx0) / fx0, (xy0[2 * i + 1] - cy0) / fy0, 1.f };
    float h1[3] = { (x1e - cx1) / fx1, (y1e - cy1) / fy1, 1.f };
    float a9[9];
#pragma unroll
    for (int k = 0; k < 9; ++k) a9[k] = h1[k / 3] * h0[k % 3];
    float w2 = w * w;
    int idx = 0;
#pragma unroll
    for (int p = 0; p < 9; ++p)
#pragma unroll
      for (int q = p; q < 9; ++q) ata[idx++] += w2 * a9[p] * a9[q];
    sumRS += rsc; sumM += rsu;
  }

  __shared__ float sh[48];
  if (threadIdx.x < 48) sh[threadIdx.x] = 0.f;
  __syncthreads();
  int lane = threadIdx.x & 31;
#pragma unroll
  for (int k = 0; k < 45; ++k) {
    float v = red32(ata[k]);
    if (lane == 0) atomicAdd(&sh[k], v);
  }
  float a = red32(sumRS), c = red32(sumM), d = red32(nkeep);
  if (lane == 0) { atomicAdd(&sh[45], a); atomicAdd(&sh[46], c); atomicAdd(&sh[47], d); }
  __syncthreads();
  if (threadIdx.x < 45) AtA[b * 45 + threadIdx.x] = sh[threadIdx.x];
  if (threadIdx.x == 45) misc[b * 4 + 0] = sh[45];
  if (threadIdx.x == 46) misc[b * 4 + 1] = sh[46];
  if (threadIdx.x == 47) misc[b * 4 + 2] = sh[47];
}

// ---------------- pass 4: tiny linear algebra + MLP + output ----------------
template <int ND>
__device__ void jacobi_sym(float S[ND][ND], float V[ND][ND], int sweeps) {
  for (int i = 0; i < ND; ++i)
    for (int j = 0; j < ND; ++j) V[i][j] = (i == j) ? 1.f : 0.f;
  for (int sw = 0; sw < sweeps; ++sw)
    for (int p = 0; p < ND - 1; ++p)
      for (int q = p + 1; q < ND; ++q) {
        float apq = S[p][q];
        if (fabsf(apq) < 1e-30f) continue;
        float th = (S[q][q] - S[p][p]) / (2.f * apq);
        float tt = (th >= 0.f ? 1.f : -1.f) / (fabsf(th) + sqrtf(th * th + 1.f));
        float cc = 1.f / sqrtf(tt * tt + 1.f);
        float ss = tt * cc;
        for (int i = 0; i < ND; ++i) { float sp = S[p][i], sq = S[q][i]; S[p][i] = cc * sp - ss * sq; S[q][i] = ss * sp + cc * sq; }
        for (int i = 0; i < ND; ++i) { float sp = S[i][p], sq = S[i][q]; S[i][p] = cc * sp - ss * sq; S[i][q] = ss * sp + cc * sq; }
        for (int i = 0; i < ND; ++i) { float vp = V[i][p], vq = V[i][q]; V[i][p] = cc * vp - ss * vq; V[i][q] = ss * vp + cc * vq; }
      }
}
__device__ __forceinline__ float det3m(const float M[3][3]) {
  return M[0][0] * (M[1][1] * M[2][2] - M[1][2] * M[2][1])
       - M[0][1] * (M[1][0] * M[2][2] - M[1][2] * M[2][0])
       + M[0][2] * (M[1][0] * M[2][1] - M[1][1] * M[2][0]);
}

__global__ void solve_kernel(const float* AtA, const float* misc, const float* sumMS,
                             const float* W1, const float* b1w, const float* W2, const float* b2w,
                             float* out, int N) {
  int b = blockIdx.x;
  if (threadIdx.x != 0) return;

  // --- min-eigenvector of AtA (9x9) -> E ---
  float S[9][9], V9[9][9];
  { int idx = 0;
    for (int p = 0; p < 9; ++p)
      for (int q = p; q < 9; ++q) { float v = AtA[b * 45 + idx++]; S[p][q] = v; S[q][p] = v; } }
  jacobi_sym<9>(S, V9, 12);
  int kmin = 0; float vmin = S[0][0];
  for (int k = 1; k < 9; ++k) if (S[k][k] < vmin) { vmin = S[k][k]; kmin = k; }
  float E[3][3];
  for (int r = 0; r < 3; ++r)
    for (int c = 0; c < 3; ++c) E[r][c] = V9[3 * r + c][kmin];

  // --- 3x3 SVD via Jacobi on E^T E ---
  float G[3][3], V3[3][3];
  for (int i = 0; i < 3; ++i)
    for (int j = 0; j < 3; ++j) {
      float s = 0.f;
      for (int k = 0; k < 3; ++k) s += E[k][i] * E[k][j];
      G[i][j] = s;
    }
  jacobi_sym<3>(G, V3, 10);
  int ord[3] = {0, 1, 2};
  for (int a = 0; a < 2; ++a)
    for (int bb = a + 1; bb < 3; ++bb)
      if (G[ord[bb]][ord[bb]] > G[ord[a]][ord[a]]) { int t = ord[a]; ord[a] = ord[bb]; ord[bb] = t; }
  float Vs[3][3], U[3][3];
  for (int k = 0; k < 3; ++k)
    for (int i = 0; i < 3; ++i) Vs[i][k] = V3[i][ord[k]];
  for (int k = 0; k < 3; ++k) {
    float u[3];
    for (int i = 0; i < 3; ++i)
      u[i] = E[i][0] * Vs[0][k] + E[i][1] * Vs[1][k] + E[i][2] * Vs[2][k];
    float nrm = sqrtf(u[0] * u[0] + u[1] * u[1] + u[2] * u[2]);
    if (nrm < 1e-12f && k == 2) {
      u[0] = U[1][0] * U[2][1] - U[2][0] * U[1][1];
      u[1] = U[2][0] * U[0][1] - U[0][0] * U[2][1];
      u[2] = U[0][0] * U[1][1] - U[1][0] * U[0][1];
      nrm = sqrtf(u[0] * u[0] + u[1] * u[1] + u[2] * u[2]) + 1e-30f;
    }
    float inn = 1.f / fmaxf(nrm, 1e-30f);
    for (int i = 0; i < 3; ++i) U[i][k] = u[i] * inn;
  }
  if (det3m(U) < 0.f)
    for (int i = 0; i < 3; ++i) for (int j = 0; j < 3; ++j) U[i][j] = -U[i][j];
  if (det3m(Vs) < 0.f)
    for (int i = 0; i < 3; ++i) for (int j = 0; j < 3; ++j) Vs[i][j] = -Vs[i][j];

  // R = U * W * Vs^T ;  UW cols: (U[:,1], -U[:,0], U[:,2])
  float UW[3][3], R[3][3];
  for (int i = 0; i < 3; ++i) { UW[i][0] = U[i][1]; UW[i][1] = -U[i][0]; UW[i][2] = U[i][2]; }
  for (int i = 0; i < 3; ++i)
    for (int j = 0; j < 3; ++j)
      R[i][j] = UW[i][0] * Vs[j][0] + UW[i][1] * Vs[j][1] + UW[i][2] * Vs[j][2];
  float t[3] = { U[0][2], U[1][2], U[2][2] };

  // --- scale head ---
  float stat0 = misc[b * 4 + 0] / (float)N;
  float stat1 = sumMS[b] / (misc[b * 4 + 1] + EPSC);
  float o = b2w[0];
  for (int j = 0; j < 32; ++j) {
    float hmid = stat0 * W1[j] + stat1 * W1[32 + j] + b1w[j];
    hmid = fmaxf(hmid, 0.f);
    o += hmid * W2[j];
  }
  float sp = (o > 20.f) ? o : log1pf(__expf(o));
  float sc = sp + 1e-6f;
  t[0] *= sc; t[1] *= sc; t[2] *= sc;

  if (misc[b * 4 + 2] < 8.f) {   // not enough correspondences -> identity pose
    for (int i = 0; i < 3; ++i) for (int j = 0; j < 3; ++j) R[i][j] = (i == j) ? 1.f : 0.f;
    t[0] = 1.f; t[1] = 0.f; t[2] = 0.f;
  }

  float* o0 = out + (size_t)b * 32;
  float* o1 = o0 + 16;
  for (int i = 0; i < 3; ++i) {
    for (int j = 0; j < 3; ++j) { o0[i * 4 + j] = R[i][j]; o1[i * 4 + j] = R[j][i]; }
    o0[i * 4 + 3] = t[i];
    o1[i * 4 + 3] = -(R[0][i] * t[0] + R[1][i] * t[1] + R[2][i] * t[2]);
  }
  for (int j = 0; j < 3; ++j) { o0[12 + j] = 0.f; o1[12 + j] = 0.f; }
  o0[15] = 1.f; o1[15] = 1.f;
}

// ---------------- host-side launch ----------------
extern "C" void kernel_launch(void* const* d_in, const int* in_sizes, int n_in,
                              void* d_out, int out_size, void* d_ws, size_t ws_size,
                              hipStream_t stream) {
  const float* feats  = (const float*)d_in[0];
  const float* scores = (const float*)d_in[1];
  const float* kpts   = (const float*)d_in[2];
  const float* Kmat   = (const float*)d_in[3];
  const float* W1     = (const float*)d_in[4];
  const float* b1w    = (const float*)d_in[5];
  const float* W2     = (const float*)d_in[6];
  const float* b2w    = (const float*)d_in[7];
  float* out = (float*)d_out;

  int B = in_sizes[3] / 18;                 // K is [B,2,3,3]
  int N = in_sizes[1] / (2 * B);            // scores [2B,N]
  int C = in_sizes[0] / in_sizes[1];        // feats  [2B,N,C]

  char* w = (char*)d_ws;
  _Float16* hfeats = (_Float16*)w;
  size_t off = (size_t)2 * B * N * C * sizeof(_Float16);
  float* rowsum = (float*)(w + off); off += (size_t)B * N * 4;
  float* colsum = (float*)(w + off); off += (size_t)B * N * 4;
  float* Mrow   = (float*)(w + off); off += (size_t)B * N * 4;
  float* Mxy    = (float*)(w + off); off += (size_t)B * N * 8;
  float* Ms1    = (float*)(w + off); off += (size_t)B * N * 4;
  float* sumMS  = (float*)(w + off); off += (size_t)B * 4;
  float* AtA    = (float*)(w + off); off += (size_t)B * 45 * 4;
  float* misc   = (float*)(w + off); off += (size_t)B * 4 * 4;

  size_t nzero = (size_t)B * N * 6 + (size_t)B * 50;
  zero_kernel<<<1024, 256, 0, stream>>>(rowsum, nzero);

  size_t ncvt = (size_t)2 * B * N * C;
  cvt_kernel<<<4096, 256, 0, stream>>>(feats, hfeats, ncvt);

  dim3 g(N / 64, B);
  sumexp_kernel<<<g, 256, 0, stream>>>(hfeats, rowsum, N, C, 0);
  sumexp_kernel<<<g, 256, 0, stream>>>(hfeats, colsum, N, C, 1);
  pass2_kernel<<<g, 256, 0, stream>>>(hfeats, rowsum, colsum, scores, kpts,
                                      Mrow, Mxy, Ms1, sumMS, N, C);
  stats_kernel<<<B, 256, 0, stream>>>(Mrow, Mxy, Ms1, scores, kpts, Kmat, AtA, misc, N);
  solve_kernel<<<B, 32, 0, stream>>>(AtA, misc, sumMS, W1, b1w, W2, b2w, out, N);
}
